// MIGANewsModel_41231686041844
// MI455X (gfx1250) — matrix operand order repaired and checked by reading, not compiled, not run on hardware
//
#include <hip/hip_runtime.h>
#include <hip/hip_bf16.h>
#include <math.h>

// ---------------------------------------------------------------------------
// MIGA news MoE model, MI455X (gfx1250) implementation.
//
// Roofline: dominated by the 512MB news_feature read (~22us @ 23.3 TB/s),
// streamed once -> loaded with non-temporal hints so it does not thrash L2.
// The only real GEMM (router: 4096x1182 @ 1182x128) runs on
// v_wmma_f32_16x16x32_f16 (f32 inputs converted to f16, f32 accumulate).
// Everything downstream is tiny per-row work (E=8 tiles) done on VALU.
// ---------------------------------------------------------------------------

typedef __attribute__((ext_vector_type(16))) _Float16 v16h;
typedef __attribute__((ext_vector_type(8)))  _Float16 v8h;
typedef __attribute__((ext_vector_type(8)))  float    v8f;
typedef __attribute__((ext_vector_type(4)))  float    v4f;

#define NROWS 4096
#define TT    32
#define DD    158
#define NEWSD 1024
#define DGRU_ 128
#define KDIM  1182      // DD + NEWSD
#define KPAD  1184      // 37 * 32, K padded for WMMA
#define GG    8
#define EE    8
#define HH    4
#define HID_  64        // G*E
#define DHH   2         // E/H

// ---------------------------------------------------------------------------
// Kernel 1: per-row aggregation. One block per row n.
//   news_agg[c] = sum_t mask[t]*news[t][c] / max(sum_t mask[t], 1e-6)
//   price_agg[c] = mean_t price[t][c]
// Writes f16 feature row [price_agg | news_agg | 0 pad] of length KPAD.
// Streaming inputs are one-touch -> non-temporal loads (TH=NT).
// ---------------------------------------------------------------------------
__global__ __launch_bounds__(256) void agg_kernel(
    const float* __restrict__ price, const float* __restrict__ news,
    const float* __restrict__ mask, _Float16* __restrict__ Xh)
{
    const int n   = blockIdx.x;
    const int tid = threadIdx.x;
    __shared__ float smask[TT];
    __shared__ float sminv;

    if (tid < TT) smask[tid] = mask[(size_t)n * TT + tid];
    __syncthreads();
    if (tid == 0) {
        float s = 0.f;
        for (int t = 0; t < TT; ++t) s += smask[t];
        sminv = 1.0f / fmaxf(s, 1e-6f);
    }
    __syncthreads();

    _Float16* xr = Xh + (size_t)n * KPAD;

    // News: 1024 columns = 256 threads * float4 each, coalesced over T.
    const v4f* np4 = (const v4f*)(news + (size_t)n * TT * NEWSD);
    v4f acc = {0.f, 0.f, 0.f, 0.f};
    for (int t = 0; t < TT; ++t) {
        const float m = smask[t];
        const v4f x = __builtin_nontemporal_load(&np4[(size_t)t * (NEWSD / 4) + tid]);
        acc += m * x;
    }
    const float inv = sminv;
    const int c = DD + tid * 4;
    xr[c + 0] = (_Float16)(acc.x * inv);
    xr[c + 1] = (_Float16)(acc.y * inv);
    xr[c + 2] = (_Float16)(acc.z * inv);
    xr[c + 3] = (_Float16)(acc.w * inv);

    // Price: 158 columns, mean over T (also one-touch streaming data).
    if (tid < DD) {
        const float* pp = price + (size_t)n * TT * DD + tid;
        float a = 0.f;
        for (int t = 0; t < TT; ++t) a += __builtin_nontemporal_load(pp + (size_t)t * DD);
        xr[tid] = (_Float16)(a * (1.0f / (float)TT));
    }
    if (tid == 0) { xr[KDIM] = (_Float16)0.f; xr[KDIM + 1] = (_Float16)0.f; }
}

// ---------------------------------------------------------------------------
// Kernel 2: transpose+convert router_W (KDIM x DGRU f32, row-major) into
// Wt (DGRU x KPAD f16, row-major, zero K-pad) so each lane's B fragment is
// two contiguous 16-byte loads.
// ---------------------------------------------------------------------------
__global__ __launch_bounds__(256) void wconv_kernel(
    const float* __restrict__ W, _Float16* __restrict__ Wt)
{
    const int idx = blockIdx.x * 256 + threadIdx.x;  // over DGRU_*KPAD
    if (idx >= DGRU_ * KPAD) return;
    const int nn = idx / KPAD;
    const int k  = idx % KPAD;
    Wt[idx] = (k < KDIM) ? (_Float16)W[(size_t)k * DGRU_ + nn] : (_Float16)0.f;
}

// ---------------------------------------------------------------------------
// Kernel 3: router GEMM h = tanh(X @ W + b) with WMMA.
// One wave (32 lanes) per 16x16 output tile; K loop of 37 steps of 32.
// Fragment layouts per CDNA5 ISA 7.12.2:
//   A (16x32 f16): lane<16 -> row=lane, K {k0..k0+7} in v0..3, {k0+16..k0+23}
//                  in v4..7; lane>=16 -> row=lane-16, K {+8..+15},{+24..+31}.
//   B (32x16 f16): lanes 0-15 -> col=lane, K {k0..k0+7} in v0..3, {k0+16..k0+23}
//                  in v4..7; lanes 16-31 -> col=lane-16, K {+8..+15},{+24..+31}.
//   C/D (16x16 f32): VGPR j, lane l -> row = j + 8*(l>>4), col = l&15.
// ---------------------------------------------------------------------------
__global__ __launch_bounds__(32) void router_wmma_kernel(
    const _Float16* __restrict__ Xh, const _Float16* __restrict__ Wt,
    const float* __restrict__ rb, float* __restrict__ hout)
{
    const int m0   = blockIdx.x * 16;
    const int n0   = blockIdx.y * 16;
    const int lane = threadIdx.x;
    const int half = lane >> 4;   // 0 or 1
    const int l15  = lane & 15;

    const _Float16* arow = Xh + (size_t)(m0 + l15) * KPAD + half * 8;
    const _Float16* brow = Wt + (size_t)(n0 + l15) * KPAD + half * 8;

    v8f c = {};
    for (int k0 = 0; k0 < KPAD; k0 += 32) {
        const v8h a0 = *(const v8h*)(arow + k0);        // K k0+half*8 .. +7
        const v8h a1 = *(const v8h*)(arow + k0 + 16);   // K k0+16+half*8 .. +7
        const v8h b0 = *(const v8h*)(brow + k0);        // K k0+half*8 .. +7
        const v8h b1 = *(const v8h*)(brow + k0 + 16);   // K k0+16+half*8 .. +7
        v16h A, B;
#pragma unroll
        for (int i = 0; i < 8; ++i) {
            A[i] = a0[i]; A[i + 8] = a1[i];
            B[i] = b0[i]; B[i + 8] = b1[i];
        }
        c = __builtin_amdgcn_wmma_f32_16x16x32_f16(
            /*neg_a=*/false, A, /*neg_b=*/false, B,
            /*c_mod=*/(short)0, c, /*reuse_a=*/false, /*reuse_b=*/false);
    }

    const int col = n0 + l15;
    const float bias = rb[col];
#pragma unroll
    for (int j = 0; j < 8; ++j) {
        const int row = m0 + half * 8 + j;
        hout[(size_t)row * DGRU_ + col] = tanhf(c[j] + bias);
    }
}

// ---------------------------------------------------------------------------
// Kernel 4: everything per-row. 64 threads per row (thread = hidden unit /
// (g,e) pair). Gate GEMV -> top-2 masked softmax -> expert GEMV -> QKV ->
// 2x2 attention per group -> output GEMV -> routing-weighted sum.
// ---------------------------------------------------------------------------
__global__ __launch_bounds__(64) void head_kernel(
    const float* __restrict__ hmat,
    const float* __restrict__ gate_W, const float* __restrict__ gate_b,
    const float* __restrict__ expert_W, const float* __restrict__ expert_b,
    const float* __restrict__ wq, const float* __restrict__ wq_b,
    const float* __restrict__ wk, const float* __restrict__ wk_b,
    const float* __restrict__ wv, const float* __restrict__ wv_b,
    const float* __restrict__ wo, const float* __restrict__ wo_b,
    float* __restrict__ pred, float* __restrict__ rw1,
    float* __restrict__ hid_out, float* __restrict__ tki,
    float* __restrict__ rw2)
{
    const int n   = blockIdx.x;
    const int tid = threadIdx.x;  // 0..63
    __shared__ float sh[DGRU_];
    __shared__ float shid[HID_];
    __shared__ float seo[HID_];
    __shared__ float sq[HID_], sk[HID_], sv[HID_], satt[HID_], sw[HID_];
    __shared__ float sM, sZ;
    __shared__ int   sidx[2];

    sh[tid]      = hmat[(size_t)n * DGRU_ + tid];
    sh[tid + 64] = hmat[(size_t)n * DGRU_ + tid + 64];
    __syncthreads();

    // hidden = h @ gate_W + gate_b
    float acc = gate_b[tid];
    for (int j = 0; j < DGRU_; ++j) acc += sh[j] * gate_W[(size_t)j * HID_ + tid];
    shid[tid] = acc;
    hid_out[(size_t)n * HID_ + tid] = acc;
    __syncthreads();

    // top-2 + masked-softmax normalizer (exp(NEG - max) underflows to 0,
    // so routing reduces to a 2-way softmax over the top-2 entries).
    if (tid == 0) {
        float v1 = -INFINITY, v2 = -INFINITY; int i1 = 0, i2 = 0;
        for (int i = 0; i < HID_; ++i) {
            const float v = shid[i];
            if (v > v1)      { v2 = v1; i2 = i1; v1 = v; i1 = i; }
            else if (v > v2) { v2 = v;  i2 = i; }
        }
        sidx[0] = i1; sidx[1] = i2;
        sM = v1;
        sZ = 1.0f + expf(v2 - v1);
        tki[(size_t)n * 2 + 0] = (float)i1;   // int indices value-cast into flat f32 out
        tki[(size_t)n * 2 + 1] = (float)i2;
    }
    __syncthreads();

    const float rw = (tid == sidx[0] || tid == sidx[1])
                         ? expf(shid[tid] - sM) / sZ : 0.0f;
    rw1[(size_t)n * HID_ + tid] = rw;
    rw2[(size_t)n * HID_ + tid] = rw;

    // expert_out[n,g,e] = hidden . expert_W[g,e,:] + expert_b[g,e]  (tid=(g,e))
    float eo = expert_b[tid];
    for (int j = 0; j < HID_; ++j) eo += shid[j] * expert_W[(size_t)tid * HID_ + j];
    seo[tid] = eo;
    __syncthreads();

    // q/k/v[n,g,f] = sum_e expert_out[n,g,e] * w{q,k,v}[g,f,e] + bias (tid=(g,f))
    {
        const int gbase = (tid >> 3) * EE;
        float aq = wq_b[tid], ak = wk_b[tid], av = wv_b[tid];
        for (int e = 0; e < EE; ++e) {
            const float x = seo[gbase + e];
            aq += x * wq[(size_t)tid * EE + e];
            ak += x * wk[(size_t)tid * EE + e];
            av += x * wv[(size_t)tid * EE + e];
        }
        sq[tid] = aq; sk[tid] = ak; sv[tid] = av;
    }
    __syncthreads();

    // Per-group attention: qt[d,h]=q[g, h*2+d]; scores 2x2; softmax over e.
    if (tid < GG) {
        const int g = tid * EE;
        const float invs = 0.70710678118654752f;  // 1/sqrt(DH=2)
        float attflat[EE];
        for (int d = 0; d < DHH; ++d) {
            float s0 = 0.f, s1 = 0.f;
            for (int hh = 0; hh < HH; ++hh) {
                const float qv = sq[g + hh * 2 + d];
                s0 += qv * sk[g + hh * 2 + 0];
                s1 += qv * sk[g + hh * 2 + 1];
            }
            s0 *= invs; s1 *= invs;
            const float mx = fmaxf(s0, s1);
            const float e0 = expf(s0 - mx), e1 = expf(s1 - mx);
            const float z  = e0 + e1;
            const float a0 = e0 / z, a1 = e1 / z;
            for (int hh = 0; hh < HH; ++hh)
                attflat[hh * 2 + d] = a0 * sv[g + hh * 2 + 0] + a1 * sv[g + hh * 2 + 1];
        }
        for (int e = 0; e < EE; ++e) satt[g + e] = attflat[e];
    }
    __syncthreads();

    // agg[n,g,f] = sum_e attended[n,g,e]*wo[g,f,e] + wo_b; weight by routing.
    {
        const int gbase = (tid >> 3) * EE;
        float ag = wo_b[tid];
        for (int e = 0; e < EE; ++e) ag += satt[gbase + e] * wo[(size_t)tid * EE + e];
        sw[tid] = ag * rw;
    }
    __syncthreads();

    if (tid == 0) {
        float s = 0.f;
        for (int i = 0; i < HID_; ++i) s += sw[i];
        pred[n] = s;
    }
}

// ---------------------------------------------------------------------------
extern "C" void kernel_launch(void* const* d_in, const int* in_sizes, int n_in,
                              void* d_out, int out_size, void* d_ws, size_t ws_size,
                              hipStream_t stream)
{
    const float* price    = (const float*)d_in[0];
    const float* news     = (const float*)d_in[1];
    const float* mask     = (const float*)d_in[2];
    const float* router_W = (const float*)d_in[3];
    const float* router_b = (const float*)d_in[4];
    const float* gate_W   = (const float*)d_in[5];
    const float* gate_b   = (const float*)d_in[6];
    const float* expert_W = (const float*)d_in[7];
    const float* expert_b = (const float*)d_in[8];
    const float* wq   = (const float*)d_in[9];
    const float* wq_b = (const float*)d_in[10];
    const float* wk   = (const float*)d_in[11];
    const float* wk_b = (const float*)d_in[12];
    const float* wv   = (const float*)d_in[13];
    const float* wv_b = (const float*)d_in[14];
    const float* wo   = (const float*)d_in[15];
    const float* wo_b = (const float*)d_in[16];

    char* ws = (char*)d_ws;
    _Float16* Xh = (_Float16*)ws;                                   // NROWS*KPAD f16
    _Float16* Wt = (_Float16*)(ws + (size_t)NROWS * KPAD * 2);      // DGRU_*KPAD f16
    float*  hmat = (float*)(ws + (size_t)NROWS * KPAD * 2
                               + (size_t)DGRU_ * KPAD * 2);         // NROWS*DGRU_ f32

    // Output tuple, flat: (pred N, rw N*64, hidden N*64, topk N*2, rw N*64)
    float* out  = (float*)d_out;
    float* pred = out;
    float* rw1  = pred + NROWS;
    float* hid  = rw1 + (size_t)NROWS * HID_;
    float* tki  = hid + (size_t)NROWS * HID_;
    float* rw2  = tki + (size_t)NROWS * 2;

    agg_kernel<<<NROWS, 256, 0, stream>>>(price, news, mask, Xh);
    wconv_kernel<<<(DGRU_ * KPAD + 255) / 256, 256, 0, stream>>>(router_W, Wt);
    router_wmma_kernel<<<dim3(NROWS / 16, DGRU_ / 16), 32, 0, stream>>>(
        Xh, Wt, router_b, hmat);
    head_kernel<<<NROWS, 64, 0, stream>>>(hmat, gate_W, gate_b,
                                          expert_W, expert_b,
                                          wq, wq_b, wk, wk_b, wv, wv_b, wo, wo_b,
                                          pred, rw1, hid, tki, rw2);
}
